// ProposalModule_11269994184880
// MI455X (gfx1250) — compile-verified, matrix operand order checked by reading, not run on hardware
//
#include <hip/hip_runtime.h>
#include <hip/hip_bf16.h>

typedef _Float16 f16;
typedef __attribute__((ext_vector_type(16))) _Float16 v16h;
typedef __attribute__((ext_vector_type(8)))  _Float16 v8h;
typedef __attribute__((ext_vector_type(8)))  float    v8f;

#define Bsz 16
#define Kpts 2048
#define Ppts 512
#define Ssz 16
#define Cch 256
#define R2 0.09f
#define PRE 256
#define HEADN 325
#define HEADPAD 336
#define TXT 10
#define MROWS (Bsz*Ppts*Ssz)   /* 131072 */
#define PROWS (Bsz*Ppts)       /* 8192 */

// ---------------------------------------------------------------- FPS
__global__ void fps_kernel(const float* __restrict__ xyz, int* __restrict__ inds) {
  __shared__ float px[Kpts], py[Kpts], pz[Kpts], dmin[Kpts];
  __shared__ float rv[256];
  __shared__ int   ri[256];
  __shared__ int   s_last;
  int b = blockIdx.x, t = threadIdx.x;
  const float* X = xyz + (size_t)b * Kpts * 3;
  for (int k = t; k < Kpts; k += 256) {
    px[k] = X[3*k]; py[k] = X[3*k+1]; pz[k] = X[3*k+2]; dmin[k] = 1e10f;
  }
  if (t == 0) { inds[b*Ppts] = 0; s_last = 0; }
  __syncthreads();
  int last = 0;
  for (int step = 1; step < Ppts; ++step) {
    float lx = px[last], ly = py[last], lz = pz[last];
    float bv = -1.0f; int bi = 0;
    for (int k = t; k < Kpts; k += 256) {
      float dx = px[k]-lx, dy = py[k]-ly, dz = pz[k]-lz;
      float d  = dx*dx + dy*dy + dz*dz;
      float dm = fminf(dmin[k], d);
      dmin[k] = dm;
      if (dm > bv) { bv = dm; bi = k; }
    }
    rv[t] = bv; ri[t] = bi;
    __syncthreads();
    for (int off = 128; off > 0; off >>= 1) {
      if (t < off) {
        float ov = rv[t+off]; int oi = ri[t+off];
        if (ov > rv[t] || (ov == rv[t] && oi < ri[t])) { rv[t] = ov; ri[t] = oi; }
      }
      __syncthreads();
    }
    if (t == 0) { s_last = ri[0]; inds[b*Ppts + step] = ri[0]; }
    __syncthreads();
    last = s_last;
  }
}

// ------------------------------------------------------- Ball query + gx tail
__global__ void ballquery_kernel(const float* __restrict__ xyz,
                                 const int* __restrict__ inds,
                                 int* __restrict__ idx,
                                 f16* __restrict__ gxt) {
  __shared__ float cx[256], cy[256], cz[256];
  int b  = blockIdx.x >> 1;
  int p  = ((blockIdx.x & 1) << 8) + threadIdx.x;
  const float* X = xyz + (size_t)b * Kpts * 3;
  int ci = inds[b*Ppts + p];
  float qx = X[3*ci], qy = X[3*ci+1], qz = X[3*ci+2];
  int base = (b*Ppts + p) * Ssz;
  int cnt = 0, first = 0;
  for (int k0 = 0; k0 < Kpts; k0 += 256) {
    int kl = k0 + threadIdx.x;
    cx[threadIdx.x] = X[3*kl]; cy[threadIdx.x] = X[3*kl+1]; cz[threadIdx.x] = X[3*kl+2];
    __syncthreads();
    for (int kk = 0; kk < 256; ++kk) {
      int k = k0 + kk;
      float dx = cx[kk]-qx, dy = cy[kk]-qy, dz = cz[kk]-qz;
      float d2 = dx*dx + dy*dy + dz*dz;
      if (d2 < R2 && cnt < Ssz) {
        if (cnt == 0) first = k;
        idx[base + cnt] = k;
        cnt++;
      }
    }
    __syncthreads();
  }
  for (int s = cnt; s < Ssz; ++s) idx[base + s] = first;
  for (int s = 0; s < Ssz; ++s) {
    int j = idx[base + s];
    float gx = (X[3*j]   - qx) * (1.0f/0.3f);
    float gy = (X[3*j+1] - qy) * (1.0f/0.3f);
    float gz = (X[3*j+2] - qz) * (1.0f/0.3f);
    f16* row = gxt + (size_t)(base + s) * 32;
    row[0] = (f16)gx; row[1] = (f16)gy; row[2] = (f16)gz;
    for (int c = 3; c < 32; ++c) row[c] = (f16)0.0f;
  }
}

// ------------------------------------------------------- weight / data prep
__global__ void prep_wt(const float* __restrict__ W, f16* __restrict__ WT,
                        int Korig, int Norig, int KdPad, int NPad, int permute) {
  int i = blockIdx.x * 256 + threadIdx.x;
  if (i >= NPad * KdPad) return;
  int n = i / KdPad, k = i % KdPad;
  float v = 0.0f;
  if (n < Norig && k < Korig) {
    int sk = k;
    if (permute) sk = (k < Korig - 3) ? (k + 3) : (k - (Korig - 3));
    v = W[(size_t)sk * Norig + n];
  }
  WT[i] = (f16)v;
}

__global__ void pad_bias(const float* __restrict__ b, float* __restrict__ o, int n, int npad) {
  int i = blockIdx.x * 256 + threadIdx.x;
  if (i < npad) o[i] = (i < n) ? b[i] : 0.0f;
}

__global__ void feats_to_f16(const float* __restrict__ F, f16* __restrict__ O) {
  int i = blockIdx.x * 256 + threadIdx.x;         // over B*K*C
  int b = i / (Kpts * Cch);
  int rem = i - b * (Kpts * Cch);
  int k = rem / Cch, c = rem - (rem / Cch) * Cch;
  O[i] = (f16)F[(size_t)b * Cch * Kpts + (size_t)c * Kpts + k];
}

// ------------------------------------------------------- WMMA helpers
__device__ inline v8f wmma_step(const f16* ap0, const f16* ap1,
                                const f16* bp0, const f16* bp1, v8f acc) {
  v8h alo = *(const v8h*)ap0, ahi = *(const v8h*)ap1;
  v8h blo = *(const v8h*)bp0, bhi = *(const v8h*)bp1;
  v16h a, b;
#pragma unroll
  for (int i = 0; i < 8; ++i) { a[i] = alo[i]; a[i+8] = ahi[i]; b[i] = blo[i]; b[i+8] = bhi[i]; }
  return __builtin_amdgcn_wmma_f32_16x16x32_f16(false, a, false, b, (short)0, acc, false, false);
}

// ------------------------------------------------------- Layer-1 gather GEMM
// A row r = (b,p,s): [ feats_kc[b, idx[r], 0:256] | gx(3) | 0-pad ]  (Kd=288)
__global__ void gemm_gather(const f16* __restrict__ F16f, const f16* __restrict__ GXT,
                            const int* __restrict__ IDX, const f16* __restrict__ W0T,
                            f16* __restrict__ OUT) {
  const int Kd = 288, N = 128;
  int wave = (blockIdx.x * blockDim.x + threadIdx.x) >> 5;
  int lane = threadIdx.x & 31;
  int tile_m = wave >> 3, tile_n = wave & 7;
  int r = tile_m * 16 + (lane & 15);
  int b = r >> 13;
  int j = IDX[r];
  const f16* arow = F16f + (((size_t)b * Kpts + j) << 8);
  const f16* grow = GXT + ((size_t)r << 5);
  const f16* brow = W0T + (size_t)(tile_n * 16 + (lane & 15)) * Kd;
  int ko = (lane >> 4) * 8;
  v8f acc = {};
#pragma unroll
  for (int kb = 0; kb < 8; ++kb) {
    int k0 = kb * 32;
    acc = wmma_step(arow + k0 + ko, arow + k0 + 16 + ko,
                    brow + k0 + ko, brow + k0 + 16 + ko, acc);
  }
  acc = wmma_step(grow + ko, grow + 16 + ko, brow + 256 + ko, brow + 256 + 16 + ko, acc);
  int cn = tile_n * 16 + (lane & 15);
#pragma unroll
  for (int v = 0; v < 8; ++v) {
    int rm = tile_m * 16 + (lane >> 4) * 8 + v;
    OUT[(size_t)rm * N + cn] = (f16)acc[v];
  }
}

// ------------------------------------------------------- generic GEMM (A MxKd f16, WT NxKd f16)
__global__ void gemm_f16(const f16* __restrict__ A, const f16* __restrict__ WT,
                         const float* __restrict__ bias,
                         f16* __restrict__ outH, float* __restrict__ outF,
                         int M, int Kd, int N, int total_tiles) {
  int wave = (blockIdx.x * blockDim.x + threadIdx.x) >> 5;
  if (wave >= total_tiles) return;              // wave-uniform: EXEC stays full
  int lane = threadIdx.x & 31;
  int tiles_n = N >> 4;
  int tile_m = wave / tiles_n, tile_n = wave - tile_m * tiles_n;
  const f16* arow = A  + (size_t)(tile_m * 16 + (lane & 15)) * Kd;
  const f16* brow = WT + (size_t)(tile_n * 16 + (lane & 15)) * Kd;
  int ko = (lane >> 4) * 8;
  v8f acc = {};
  for (int k0 = 0; k0 < Kd; k0 += 32)
    acc = wmma_step(arow + k0 + ko, arow + k0 + 16 + ko,
                    brow + k0 + ko, brow + k0 + 16 + ko, acc);
  int cn = tile_n * 16 + (lane & 15);
  float bv = bias ? bias[cn] : 0.0f;
#pragma unroll
  for (int v = 0; v < 8; ++v) {
    int rm = tile_m * 16 + (lane >> 4) * 8 + v;
    float val = acc[v] + bv;
    if (outH) outH[(size_t)rm * N + cn] = (f16)val;
    if (outF) outF[(size_t)rm * N + cn] = val;
  }
}

// ------------------------------------------------------- batch-norm stats / apply
__global__ void colstats(const f16* __restrict__ Y, float* __restrict__ stats, int M, int N) {
  __shared__ float s1[256], s2[256];
  int c = blockIdx.x, t = threadIdx.x;
  float a = 0.0f, b = 0.0f;
  for (int r = t; r < M; r += 256) {
    float v = (float)Y[(size_t)r * N + c];
    a += v; b += v * v;
  }
  s1[t] = a; s2[t] = b;
  __syncthreads();
  for (int off = 128; off > 0; off >>= 1) {
    if (t < off) { s1[t] += s1[t+off]; s2[t] += s2[t+off]; }
    __syncthreads();
  }
  if (t == 0) { stats[c] = s1[0]; stats[N + c] = s2[0]; }
}

__global__ void bn_apply(f16* __restrict__ Y, const float* __restrict__ stats,
                         const float* __restrict__ gam, const float* __restrict__ bet,
                         int M, int N) {
  int i = blockIdx.x * 256 + threadIdx.x;
  if (i >= M * N) return;
  int c = i % N;
  float invM = 1.0f / (float)M;
  float mean = stats[c] * invM;
  float var  = stats[N + c] * invM - mean * mean;
  float sc = gam[c] * rsqrtf(var + 1e-5f);
  float sh = bet[c] - mean * sc;
  float v = (float)Y[i] * sc + sh;
  Y[i] = (f16)fmaxf(v, 0.0f);
}

// ------------------------------------------------------- max over S, clip extract
__global__ void smax16(const f16* __restrict__ H, f16* __restrict__ FEAT) {
  int i = blockIdx.x * 256 + threadIdx.x;         // over PROWS*128
  int g = i >> 7, c = i & 127;
  const f16* base = H + (size_t)g * (Ssz * 128) + c;
  float m = -1e30f;
#pragma unroll
  for (int s = 0; s < Ssz; ++s) m = fmaxf(m, (float)base[s * 128]);
  FEAT[i] = (f16)m;
}

__global__ void clip_extract(const float* __restrict__ NET3, f16* __restrict__ CLIP) {
  int i = blockIdx.x * 256 + threadIdx.x;         // over PROWS*PRE
  int r = i >> 8, c = i & 255;
  CLIP[i] = (f16)NET3[(size_t)r * HEADPAD + 69 + c];
}

// ------------------------------------------------------- normalization + logits
__global__ void norm_text(const float* __restrict__ T, float* __restrict__ TN) {
  __shared__ float red[128];
  int r = blockIdx.x, t = threadIdx.x;
  float ss = 0.0f;
  for (int c = t; c < 512; c += 128) { float v = T[r*512 + c]; ss += v * v; }
  red[t] = ss; __syncthreads();
  for (int off = 64; off > 0; off >>= 1) { if (t < off) red[t] += red[t+off]; __syncthreads(); }
  float rinv = rsqrtf(red[0]);
  for (int c = t; c < 512; c += 128) TN[r*512 + c] = T[r*512 + c] * rinv;
}

__global__ void qnorm_logits(const float* __restrict__ Q, const float* __restrict__ TN,
                             float* __restrict__ LOG) {
  __shared__ float sq[512];
  __shared__ float red[128];
  int r = blockIdx.x, t = threadIdx.x;
  const float* q = Q + (size_t)r * 512;
  float ss = 0.0f;
  for (int c = t; c < 512; c += 128) { float v = q[c]; sq[c] = v; ss += v * v; }
  red[t] = ss; __syncthreads();
  for (int off = 64; off > 0; off >>= 1) { if (t < off) red[t] += red[t+off]; __syncthreads(); }
  float rinv = rsqrtf(red[0]);
  __syncthreads();
  for (int tt = 0; tt < TXT; ++tt) {
    float d = 0.0f;
    for (int c = t; c < 512; c += 128) d += sq[c] * TN[tt*512 + c];
    red[t] = d; __syncthreads();
    for (int off = 64; off > 0; off >>= 1) { if (t < off) red[t] += red[t+off]; __syncthreads(); }
    if (t == 0) LOG[r*TXT + tt] = red[0] * rinv * (1.0f / 0.07f);
    __syncthreads();
  }
}

// ------------------------------------------------------- output assembly (B,79,P)
__global__ void out_assemble(const float* __restrict__ NET3, const float* __restrict__ LOG,
                             float* __restrict__ out) {
  int i = blockIdx.x * 256 + threadIdx.x;         // over 16*79*512
  if (i >= Bsz * 79 * Ppts) return;
  int b = i / (79 * Ppts);
  int c = (i / Ppts) % 79;
  int p = i % Ppts;
  int r = b * Ppts + p;
  out[i] = (c < 69) ? NET3[(size_t)r * HEADPAD + c] : LOG[r * TXT + (c - 69)];
}

// ================================================================ host side
static inline size_t alignup(size_t x) { return (x + 255) & ~(size_t)255; }

extern "C" void kernel_launch(void* const* d_in, const int* in_sizes, int n_in,
                              void* d_out, int out_size, void* d_ws, size_t ws_size,
                              hipStream_t stream) {
  const float* xyz      = (const float*)d_in[0];
  const float* features = (const float*)d_in[1];
  const float* textf    = (const float*)d_in[2];
  const float* g_w0 = (const float*)d_in[3];
  const float* g_w1 = (const float*)d_in[4];
  const float* g_w2 = (const float*)d_in[5];
  const float* g_gam0 = (const float*)d_in[6],  *g_bet0 = (const float*)d_in[7];
  const float* g_gam1 = (const float*)d_in[8],  *g_bet1 = (const float*)d_in[9];
  const float* g_gam2 = (const float*)d_in[10], *g_bet2 = (const float*)d_in[11];
  const float* conv1_w = (const float*)d_in[12], *conv1_b = (const float*)d_in[13];
  const float* bn1_g = (const float*)d_in[14],   *bn1_b = (const float*)d_in[15];
  const float* conv2_w = (const float*)d_in[16], *conv2_b = (const float*)d_in[17];
  const float* bn2_g = (const float*)d_in[18],   *bn2_b = (const float*)d_in[19];
  const float* conv3_w = (const float*)d_in[20], *conv3_b = (const float*)d_in[21];
  const float* h_w0 = (const float*)d_in[22], *h_b0 = (const float*)d_in[23];
  const float* h_g0 = (const float*)d_in[24], *h_be0 = (const float*)d_in[25];
  const float* h_w1 = (const float*)d_in[26], *h_b1 = (const float*)d_in[27];
  const float* h_g1 = (const float*)d_in[28], *h_be1 = (const float*)d_in[29];
  const float* h_w2 = (const float*)d_in[30], *h_b2 = (const float*)d_in[31];

  char* ws = (char*)d_ws;
  size_t o = 0;
  auto take = [&](size_t bytes) { char* p = ws + o; o = alignup(o + bytes); return p; };

  int*   INDS = (int*)take(PROWS * 4);
  int*   IDX  = (int*)take(MROWS * 4);
  f16*   F16F = (f16*)take((size_t)Bsz * Kpts * Cch * 2);
  f16*   GXT  = (f16*)take((size_t)MROWS * 32 * 2);
  f16*   W0T  = (f16*)take(128 * 288 * 2);
  f16*   W1T  = (f16*)take(128 * 128 * 2);
  f16*   W2T  = (f16*)take(128 * 128 * 2);
  f16*   C1T  = (f16*)take(128 * 128 * 2);
  f16*   C2T  = (f16*)take(128 * 128 * 2);
  f16*   C3T  = (f16*)take(HEADPAD * 128 * 2);
  f16*   H0T  = (f16*)take(512 * 256 * 2);
  f16*   H1TW = (f16*)take(512 * 512 * 2);
  f16*   H2TW = (f16*)take(512 * 512 * 2);
  float* B3P  = (float*)take(HEADPAD * 4);
  f16*   H1   = (f16*)take((size_t)MROWS * 128 * 2);
  f16*   H2   = (f16*)take((size_t)MROWS * 128 * 2);
  float* ST   = (float*)take(2 * 512 * 4);
  f16*   FEAT = (f16*)take((size_t)PROWS * 128 * 2);
  f16*   N1   = (f16*)take((size_t)PROWS * 128 * 2);
  f16*   N2   = (f16*)take((size_t)PROWS * 128 * 2);
  float* NET3 = (float*)take((size_t)PROWS * HEADPAD * 4);
  f16*   CLIP = (f16*)take((size_t)PROWS * PRE * 2);
  f16*   HH1  = (f16*)take((size_t)PROWS * 512 * 2);
  f16*   HH2  = (f16*)take((size_t)PROWS * 512 * 2);
  float* Q    = (float*)take((size_t)PROWS * 512 * 4);
  float* TN   = (float*)take(TXT * 512 * 4);
  float* LOG  = (float*)take((size_t)PROWS * TXT * 4);
  (void)ws_size; (void)n_in; (void)in_sizes; (void)out_size;

  // ---- prep: weights (transposed to NxK f16), features (b,k,c) f16
  prep_wt<<<(128*288+255)/256, 256, 0, stream>>>(g_w0, W0T, 259, 128, 288, 128, 1);
  prep_wt<<<64, 256, 0, stream>>>(g_w1, W1T, 128, 128, 128, 128, 0);
  prep_wt<<<64, 256, 0, stream>>>(g_w2, W2T, 128, 128, 128, 128, 0);
  prep_wt<<<64, 256, 0, stream>>>(conv1_w, C1T, 128, 128, 128, 128, 0);
  prep_wt<<<64, 256, 0, stream>>>(conv2_w, C2T, 128, 128, 128, 128, 0);
  prep_wt<<<(HEADPAD*128+255)/256, 256, 0, stream>>>(conv3_w, C3T, 128, HEADN, 128, HEADPAD, 0);
  prep_wt<<<512, 256, 0, stream>>>(h_w0, H0T, 256, 512, 256, 512, 0);
  prep_wt<<<1024, 256, 0, stream>>>(h_w1, H1TW, 512, 512, 512, 512, 0);
  prep_wt<<<1024, 256, 0, stream>>>(h_w2, H2TW, 512, 512, 512, 512, 0);
  pad_bias<<<2, 256, 0, stream>>>(conv3_b, B3P, HEADN, HEADPAD);
  feats_to_f16<<<(Bsz*Kpts*Cch)/256, 256, 0, stream>>>(features, F16F);

  // ---- FPS + ball query + grouping
  fps_kernel<<<Bsz, 256, 0, stream>>>(xyz, INDS);
  ballquery_kernel<<<Bsz*2, 256, 0, stream>>>(xyz, INDS, IDX, GXT);

  // ---- grouped MLP: 259->128->128->128, bn+relu each, max over S
  gemm_gather<<<(MROWS/16)*8/8, 256, 0, stream>>>(F16F, GXT, IDX, W0T, H1);
  colstats<<<128, 256, 0, stream>>>(H1, ST, MROWS, 128);
  bn_apply<<<MROWS*128/256, 256, 0, stream>>>(H1, ST, g_gam0, g_bet0, MROWS, 128);

  gemm_f16<<<(MROWS/16)*8/8, 256, 0, stream>>>(H1, W1T, nullptr, H2, nullptr, MROWS, 128, 128, (MROWS/16)*8);
  colstats<<<128, 256, 0, stream>>>(H2, ST, MROWS, 128);
  bn_apply<<<MROWS*128/256, 256, 0, stream>>>(H2, ST, g_gam1, g_bet1, MROWS, 128);

  gemm_f16<<<(MROWS/16)*8/8, 256, 0, stream>>>(H2, W2T, nullptr, H1, nullptr, MROWS, 128, 128, (MROWS/16)*8);
  colstats<<<128, 256, 0, stream>>>(H1, ST, MROWS, 128);
  bn_apply<<<MROWS*128/256, 256, 0, stream>>>(H1, ST, g_gam2, g_bet2, MROWS, 128);

  smax16<<<PROWS*128/256, 256, 0, stream>>>(H1, FEAT);

  // ---- proposal head convs
  gemm_f16<<<(PROWS/16)*8/8, 256, 0, stream>>>(FEAT, C1T, conv1_b, N1, nullptr, PROWS, 128, 128, (PROWS/16)*8);
  colstats<<<128, 256, 0, stream>>>(N1, ST, PROWS, 128);
  bn_apply<<<PROWS*128/256, 256, 0, stream>>>(N1, ST, bn1_g, bn1_b, PROWS, 128);

  gemm_f16<<<(PROWS/16)*8/8, 256, 0, stream>>>(N1, C2T, conv2_b, N2, nullptr, PROWS, 128, 128, (PROWS/16)*8);
  colstats<<<128, 256, 0, stream>>>(N2, ST, PROWS, 128);
  bn_apply<<<PROWS*128/256, 256, 0, stream>>>(N2, ST, bn2_g, bn2_b, PROWS, 128);

  gemm_f16<<<(PROWS/16)*(HEADPAD/16)/8, 256, 0, stream>>>(N2, C3T, B3P, nullptr, NET3, PROWS, 128, HEADPAD, (PROWS/16)*(HEADPAD/16));

  // ---- CLIP head: 256->512->512->512
  clip_extract<<<PROWS*PRE/256, 256, 0, stream>>>(NET3, CLIP);

  gemm_f16<<<(PROWS/16)*32/8, 256, 0, stream>>>(CLIP, H0T, h_b0, HH1, nullptr, PROWS, 256, 512, (PROWS/16)*32);
  colstats<<<512, 256, 0, stream>>>(HH1, ST, PROWS, 512);
  bn_apply<<<PROWS*512/256, 256, 0, stream>>>(HH1, ST, h_g0, h_be0, PROWS, 512);

  gemm_f16<<<(PROWS/16)*32/8, 256, 0, stream>>>(HH1, H1TW, h_b1, HH2, nullptr, PROWS, 512, 512, (PROWS/16)*32);
  colstats<<<512, 256, 0, stream>>>(HH2, ST, PROWS, 512);
  bn_apply<<<PROWS*512/256, 256, 0, stream>>>(HH2, ST, h_g1, h_be1, PROWS, 512);

  gemm_f16<<<(PROWS/16)*32/8, 256, 0, stream>>>(HH2, H2TW, h_b2, nullptr, Q, PROWS, 512, 512, (PROWS/16)*32);

  // ---- cosine logits + output
  norm_text<<<TXT, 128, 0, stream>>>(textf, TN);
  qnorm_logits<<<PROWS, 128, 0, stream>>>(Q, TN, LOG);
  out_assemble<<<(Bsz*79*Ppts + 255)/256, 256, 0, stream>>>(NET3, LOG, (float*)d_out);
}
